// GRUDecoder_2052994367646
// MI455X (gfx1250) — matrix operand order repaired and checked by reading, not compiled, not run on hardware
//
#include <hip/hip_runtime.h>
#include <hip/hip_bf16.h>

// ---------------- problem constants ----------------
#define B_    16
#define T_    40
#define N_    64
#define DIN_  64
#define H_    256
#define E_    (N_ * (N_ - 1))   // 4032
#define DOUT_ 64
#define TS_   (T_ - 1)          // 39

typedef _Float16 h16;
typedef __attribute__((ext_vector_type(8)))  _Float16 v8h;
typedef __attribute__((ext_vector_type(16))) _Float16 v16h;
typedef __attribute__((ext_vector_type(8)))  float    v8f;

// ---------------- WMMA fragment helpers (CDNA5 wave32 layouts) ----------------
// A: 16x32 f16 (MxK). lane L: row M=L%16; VGPRs 0-3 hold K = kb+{0..7},
// VGPRs 4-7 hold K = 16+kb+{0..7}, kb = (L<16)?0:8  -> two contiguous v8h loads.
static __device__ __forceinline__ v16h load_a_frag(const h16* base, int stride,
                                                   int row0, int k0) {
  const int lane = threadIdx.x & 31;
  const int row  = row0 + (lane & 15);
  const int kb   = (lane < 16) ? 0 : 8;
  const h16* p = base + row * stride + k0 + kb;
  v8h lo = *(const v8h*)p;
  v8h hi = *(const v8h*)(p + 16);
  v16h a;
#pragma unroll
  for (int i = 0; i < 8; ++i) { a[i] = lo[i]; a[i + 8] = hi[i]; }
  return a;
}

// B: 32x16 f16 (KxN). lane L: col N=L%16; elems 0-7 K = kb+{0..7},
// elems 8-15 K = 16+kb+{0..7}, kb=(L<16)?0:8 (weights stored [fout][fin] row-major).
static __device__ __forceinline__ v16h load_b_frag(const h16* W, int fin,
                                                   int k0, int n0) {
  const int lane = threadIdx.x & 31;
  const int n  = n0 + (lane & 15);
  const int kb = (lane < 16) ? 0 : 8;
  const h16* p = W + n * fin + k0 + kb;
  v8h lo = *(const v8h*)p;
  v8h hi = *(const v8h*)(p + 16);
  v16h b;
#pragma unroll
  for (int i = 0; i < 8; ++i) { b[i] = lo[i]; b[i + 8] = hi[i]; }
  return b;
}

static __device__ __forceinline__ v8f wmma16(v16h a, v16h b, v8f c) {
  return __builtin_amdgcn_wmma_f32_16x16x32_f16(false, a, false, b,
                                                (short)0, c, false, false);
}

static __device__ __forceinline__ float sigmoidf_(float x) {
  return 1.0f / (1.0f + __expf(-x));
}

// CDNA5 async global->LDS copy (ASYNCcnt path). ldsOff = wave-relative LDS
// byte address in a VGPR (VDST), gaddr = 64-bit global address (VADDR), GV mode.
static __device__ __forceinline__ void async_copy_b128(unsigned ldsOff,
                                                       const void* gaddr) {
  asm volatile("global_load_async_to_lds_b128 %0, %1, off"
               :: "v"(ldsOff), "v"((unsigned long long)(uintptr_t)gaddr)
               : "memory");
}
static __device__ __forceinline__ void async_wait0() {
  asm volatile("s_wait_asynccnt 0" ::: "memory");
}

// ---------------- prep kernels ----------------
__global__ void cvt_f32_f16_kernel(const float* __restrict__ s,
                                   h16* __restrict__ d, int n) {
  int i = blockIdx.x * blockDim.x + threadIdx.x;
  if (i < n) d[i] = (h16)s[i];
}

__global__ void tsum_kernel(const float* __restrict__ rt,
                            float* __restrict__ ts, int n) {
  int i = blockIdx.x * blockDim.x + threadIdx.x;
  if (i < n) ts[i] = rt[2 * i] + rt[2 * i + 1];
}

// ---------------- Kernel A: message MLP + type mask + edge2node ----------------
// One workgroup per (batch b, receiver node j): 63 incoming edges (+1 pad row).
// Row r = sender i (i != j): pre_msg row = [h[b,i,:], h[b,j,:]].
// Waves partition the 16 output-column tiles (4 each) and sweep all 4 row
// tiles, so every W fragment is loaded from global exactly once per workgroup
// and reused across 4 WMMAs in registers.
__global__ __launch_bounds__(128) void msg_agg_kernel(
    const h16* __restrict__ hid16,   // [B,N,H]  f16 hidden
    const float* __restrict__ tsum,  // [B,E]    rel_types[...,0]+rel_types[...,1]
    const h16* __restrict__ W1,      // [H, 2H]
    const float* __restrict__ b1,    // [H]
    const h16* __restrict__ W2,      // [H, H]
    const float* __restrict__ b2,    // [H]
    float* __restrict__ agg)         // [B,N,H]  aggregated messages
{
  extern __shared__ char smem[];
  h16*   preA = (h16*)smem;                               // 64 x 512 f16
  h16*   mid  = (h16*)(smem + 64 * 512 * 2);              // 64 x 256 f16
  float* tsr  = (float*)(smem + 64 * 512 * 2 + 64 * 256 * 2); // 64 f32

  const int wg   = blockIdx.x;
  const int b    = wg >> 6;
  const int j    = wg & 63;
  const int tid  = threadIdx.x;
  const int lane = tid & 31;
  const int wave = tid >> 5;

  if (tid < 64) {
    float t = 0.0f;
    if (tid < 63) {
      int i = (tid < j) ? tid : tid + 1;              // sender
      int e = i * 63 + ((j < i) ? j : j - 1);         // edge id (i -> j)
      t = tsum[b * E_ + e];
    }
    tsr[tid] = t;                                     // pad row -> 0
  }

  // ---- async-gather pre_msg rows into LDS (ASYNCcnt) ----
  {
    const h16* hb = hid16 + (size_t)b * N_ * H_;
    const unsigned ldsBase = (unsigned)(uintptr_t)preA; // low 32 bits = LDS addr
    for (int c = tid; c < 64 * 64; c += 128) {        // 16B chunks: 64 rows x 64
      int row = c >> 6;
      int cc  = (c & 63) * 8;                         // half offset in row
      int i = (row < j) ? row : row + 1;
      if (row == 63) i = j;                           // pad row (masked by ts=0)
      const h16* src = (cc < H_) ? (hb + i * H_ + cc) : (hb + j * H_ + (cc - H_));
      async_copy_b128(ldsBase + (unsigned)(row * 1024 + cc * 2), src);
    }
    async_wait0();
  }
  __syncthreads();

  const int nl  = lane & 15;
  const int grh = (lane >> 4) << 3;                   // 0 or 8 (C row half)

  // fc1: [64x512] x [512x256] -> relu -> mid (f16); wave owns nt = 4w..4w+3
  for (int ni = 0; ni < 4; ++ni) {
    const int nt = wave * 4 + ni;
    v8f acc0 = {}, acc1 = {}, acc2 = {}, acc3 = {};
#pragma unroll
    for (int kt = 0; kt < 16; ++kt) {
      v16h bw = load_b_frag(W1, 512, kt * 32, nt * 16);
      acc0 = wmma16(load_a_frag(preA, 512,  0, kt * 32), bw, acc0);
      acc1 = wmma16(load_a_frag(preA, 512, 16, kt * 32), bw, acc1);
      acc2 = wmma16(load_a_frag(preA, 512, 32, kt * 32), bw, acc2);
      acc3 = wmma16(load_a_frag(preA, 512, 48, kt * 32), bw, acc3);
    }
    const int n = nt * 16 + nl;
    const float bias = b1[n];
#pragma unroll
    for (int g = 0; g < 8; ++g) {
      float v0 = acc0[g] + bias, v1 = acc1[g] + bias;
      float v2 = acc2[g] + bias, v3 = acc3[g] + bias;
      mid[( 0 + grh + g) * H_ + n] = (h16)(v0 > 0.0f ? v0 : 0.0f);
      mid[(16 + grh + g) * H_ + n] = (h16)(v1 > 0.0f ? v1 : 0.0f);
      mid[(32 + grh + g) * H_ + n] = (h16)(v2 > 0.0f ? v2 : 0.0f);
      mid[(48 + grh + g) * H_ + n] = (h16)(v3 > 0.0f ? v3 : 0.0f);
    }
  }
  __syncthreads();

  // fc2: [64x256] x [256x256] -> relu -> *type_sum -> column sum -> agg (global)
  float* ag = agg + ((size_t)b * N_ + j) * H_;
  for (int ni = 0; ni < 4; ++ni) {
    const int nt = wave * 4 + ni;
    v8f acc0 = {}, acc1 = {}, acc2 = {}, acc3 = {};
#pragma unroll
    for (int kt = 0; kt < 8; ++kt) {
      v16h bw = load_b_frag(W2, H_, kt * 32, nt * 16);
      acc0 = wmma16(load_a_frag(mid, H_,  0, kt * 32), bw, acc0);
      acc1 = wmma16(load_a_frag(mid, H_, 16, kt * 32), bw, acc1);
      acc2 = wmma16(load_a_frag(mid, H_, 32, kt * 32), bw, acc2);
      acc3 = wmma16(load_a_frag(mid, H_, 48, kt * 32), bw, acc3);
    }
    const int n = nt * 16 + nl;
    const float bias = b2[n];
    float part = 0.0f;
#pragma unroll
    for (int g = 0; g < 8; ++g) {
      float v0 = acc0[g] + bias; v0 = v0 > 0.0f ? v0 : 0.0f;
      float v1 = acc1[g] + bias; v1 = v1 > 0.0f ? v1 : 0.0f;
      float v2 = acc2[g] + bias; v2 = v2 > 0.0f ? v2 : 0.0f;
      float v3 = acc3[g] + bias; v3 = v3 > 0.0f ? v3 : 0.0f;
      part += v0 * tsr[ 0 + grh + g] + v1 * tsr[16 + grh + g]
            + v2 * tsr[32 + grh + g] + v3 * tsr[48 + grh + g];
    }
    part += __shfl_xor(part, 16, 32);                 // combine C row halves
    if (lane < 16) ag[n] = part;                      // each n owned by one wave
  }
}

// ---------------- Kernel B: GRU update + 3-layer output MLP ----------------
// One workgroup per 16 contiguous (b,n) rows; 4 waves split the 256 output cols.
__global__ __launch_bounds__(128) void gru_out_kernel(
    const h16* __restrict__ x16,     // [B,T,N,DIN] f16 inputs
    int t,
    const float* __restrict__ agg,   // [B,N,H]
    float* __restrict__ h32,         // [B,N,H]  f32 hidden (in/out)
    h16* __restrict__ hid16,         // [B,N,H]  f16 hidden (out)
    const h16* __restrict__ Wir, const h16* __restrict__ Wii, const h16* __restrict__ Win,
    const h16* __restrict__ Whr, const h16* __restrict__ Whi, const h16* __restrict__ Whn,
    const h16* __restrict__ Wo1, const h16* __restrict__ Wo2, const h16* __restrict__ Wo3,
    const float* __restrict__ bir, const float* __restrict__ bii, const float* __restrict__ bin_,
    const float* __restrict__ bhr, const float* __restrict__ bhi, const float* __restrict__ bhn,
    const float* __restrict__ bo1, const float* __restrict__ bo2, const float* __restrict__ bo3,
    float* __restrict__ out)         // [B,T-1,N,DOUT]
{
  extern __shared__ char smem[];
  h16* xs = (h16*)smem;              // 16 x 64
  h16* as = xs + 16 * 64;            // 16 x 256
  h16* hs = as + 16 * 256;           // 16 x 256 (new hidden)
  h16* p1 = hs + 16 * 256;           // 16 x 256
  h16* p2 = p1 + 16 * 256;           // 16 x 256

  const int tid  = threadIdx.x;
  const int lane = tid & 31;
  const int wave = tid >> 5;
  const int m0   = blockIdx.x * 16;  // global row base (b*64+n)
  const int b    = m0 >> 6;
  const int nrow0 = m0 & 63;

  {                                   // stage x rows for this timestep
    int row = tid >> 3;
    int cc  = (tid & 7) * 8;
    const h16* src = x16 + (((size_t)b * T_ + t) * N_ + nrow0 + row) * DIN_ + cc;
    *(v8h*)(xs + row * DIN_ + cc) = *(const v8h*)src;
  }
  for (int c = tid; c < 16 * H_; c += 128) {          // agg f32 -> f16
    int row = c >> 8, cc = c & 255;
    as[row * H_ + cc] = (h16)agg[(size_t)(m0 + row) * H_ + cc];
  }
  __syncthreads();

  // GRU gates: r,i,n  (x-path K=64, agg-path K=256)
  for (int ntb = 0; ntb < 4; ++ntb) {
    const int nt = wave + ntb * 4;
    v8f aR = {}, aI = {}, aN = {}, aH = {};
#pragma unroll
    for (int kt = 0; kt < 2; ++kt) {
      v16h a = load_a_frag(xs, DIN_, 0, kt * 32);
      aR = wmma16(a, load_b_frag(Wir, DIN_, kt * 32, nt * 16), aR);
      aI = wmma16(a, load_b_frag(Wii, DIN_, kt * 32, nt * 16), aI);
      aN = wmma16(a, load_b_frag(Win, DIN_, kt * 32, nt * 16), aN);
    }
#pragma unroll
    for (int kt = 0; kt < 8; ++kt) {
      v16h a = load_a_frag(as, H_, 0, kt * 32);
      aR = wmma16(a, load_b_frag(Whr, H_, kt * 32, nt * 16), aR);
      aI = wmma16(a, load_b_frag(Whi, H_, kt * 32, nt * 16), aI);
      aH = wmma16(a, load_b_frag(Whn, H_, kt * 32, nt * 16), aH);
    }
    const int n  = nt * 16 + (lane & 15);
    const int gr = (lane >> 4) << 3;
    const float cbr = bir[n] + bhr[n];
    const float cbi = bii[n] + bhi[n];
    const float cbn = bin_[n];
    const float cbh = bhn[n];
#pragma unroll
    for (int g = 0; g < 8; ++g) {
      float r  = sigmoidf_(aR[g] + cbr);
      float ii = sigmoidf_(aI[g] + cbi);
      float nn = tanhf(aN[g] + cbn + r * (aH[g] + cbh));
      size_t idx = (size_t)(m0 + gr + g) * H_ + n;
      float hold = h32[idx];
      float hnew = (1.0f - ii) * nn + ii * hold;
      h32[idx] = hnew;
      hid16[idx] = (h16)hnew;
      hs[(gr + g) * H_ + n] = (h16)hnew;
    }
  }
  __syncthreads();

  // out_fc1
  for (int ntb = 0; ntb < 4; ++ntb) {
    const int nt = wave + ntb * 4;
    v8f acc = {};
#pragma unroll
    for (int kt = 0; kt < 8; ++kt)
      acc = wmma16(load_a_frag(hs, H_, 0, kt * 32),
                   load_b_frag(Wo1, H_, kt * 32, nt * 16), acc);
    const int n = nt * 16 + (lane & 15);
    const float bias = bo1[n];
    const int gr = (lane >> 4) << 3;
#pragma unroll
    for (int g = 0; g < 8; ++g) {
      float v = acc[g] + bias;
      p1[(gr + g) * H_ + n] = (h16)(v > 0.0f ? v : 0.0f);
    }
  }
  __syncthreads();

  // out_fc2
  for (int ntb = 0; ntb < 4; ++ntb) {
    const int nt = wave + ntb * 4;
    v8f acc = {};
#pragma unroll
    for (int kt = 0; kt < 8; ++kt)
      acc = wmma16(load_a_frag(p1, H_, 0, kt * 32),
                   load_b_frag(Wo2, H_, kt * 32, nt * 16), acc);
    const int n = nt * 16 + (lane & 15);
    const float bias = bo2[n];
    const int gr = (lane >> 4) << 3;
#pragma unroll
    for (int g = 0; g < 8; ++g) {
      float v = acc[g] + bias;
      p2[(gr + g) * H_ + n] = (h16)(v > 0.0f ? v : 0.0f);
    }
  }
  __syncthreads();

  // out_fc3 (N=64): one 16-col tile per wave
  {
    const int nt = wave;
    v8f acc = {};
#pragma unroll
    for (int kt = 0; kt < 8; ++kt)
      acc = wmma16(load_a_frag(p2, H_, 0, kt * 32),
                   load_b_frag(Wo3, H_, kt * 32, nt * 16), acc);
    const int n = nt * 16 + (lane & 15);
    const float bias = bo3[n];
    const int gr = (lane >> 4) << 3;
#pragma unroll
    for (int g = 0; g < 8; ++g) {
      size_t o = (((size_t)b * TS_ + t) * N_ + (nrow0 + gr + g)) * DOUT_ + n;
      out[o] = acc[g] + bias;
    }
  }
}

// ---------------- host launch ----------------
// workspace layout (bytes, 256-aligned)
#define OFF_W1  0u          // 256*512 f16
#define OFF_W2  262144u     // 256*256 f16
#define OFF_WIR 393216u
#define OFF_WII 425984u
#define OFF_WIN 458752u
#define OFF_WHR 491520u
#define OFF_WHI 622592u
#define OFF_WHN 753664u
#define OFF_WO1 884736u
#define OFF_WO2 1015808u
#define OFF_WO3 1146880u
#define OFF_X16 1179648u    // B*T*N*DIN f16
#define OFF_TS  6422528u    // B*E f32
#define OFF_H32 6680576u    // B*N*H f32
#define OFF_H16 7729152u    // B*N*H f16
#define OFF_AGG 8253440u    // B*N*H f32

#define SMEM_A (64 * 512 * 2 + 64 * 256 * 2 + 64 * 4)   // 98560 B
#define SMEM_B ((16 * 64 + 4 * 16 * 256) * 2)           // 34816 B

extern "C" void kernel_launch(void* const* d_in, const int* in_sizes, int n_in,
                              void* d_out, int out_size, void* d_ws, size_t ws_size,
                              hipStream_t stream) {
  (void)in_sizes; (void)n_in; (void)out_size; (void)ws_size;
  char* ws = (char*)d_ws;

  const float* inputs    = (const float*)d_in[0];
  const float* rel_types = (const float*)d_in[3];

  h16* W1h  = (h16*)(ws + OFF_W1);
  h16* W2h  = (h16*)(ws + OFF_W2);
  h16* Wir  = (h16*)(ws + OFF_WIR);
  h16* Wii  = (h16*)(ws + OFF_WII);
  h16* Win  = (h16*)(ws + OFF_WIN);
  h16* Whr  = (h16*)(ws + OFF_WHR);
  h16* Whi  = (h16*)(ws + OFF_WHI);
  h16* Whn  = (h16*)(ws + OFF_WHN);
  h16* Wo1  = (h16*)(ws + OFF_WO1);
  h16* Wo2  = (h16*)(ws + OFF_WO2);
  h16* Wo3  = (h16*)(ws + OFF_WO3);
  h16* x16  = (h16*)(ws + OFF_X16);
  float* ts = (float*)(ws + OFF_TS);
  float* h32 = (float*)(ws + OFF_H32);
  h16* h16p = (h16*)(ws + OFF_H16);
  float* agg = (float*)(ws + OFF_AGG);

  // allow >64KB dynamic LDS (320KB/WGP on gfx1250)
  hipFuncSetAttribute((const void*)msg_agg_kernel,
                      hipFuncAttributeMaxDynamicSharedMemorySize, SMEM_A);
  hipFuncSetAttribute((const void*)gru_out_kernel,
                      hipFuncAttributeMaxDynamicSharedMemorySize, SMEM_B);

  // weight / input f32 -> f16 conversions
  struct { const float* s; h16* d; int n; } cv[12] = {
    { (const float*)d_in[4],  W1h, H_ * 2 * H_ },
    { (const float*)d_in[6],  W2h, H_ * H_ },
    { (const float*)d_in[8],  Wir, H_ * DIN_ },
    { (const float*)d_in[12], Wii, H_ * DIN_ },
    { (const float*)d_in[16], Win, H_ * DIN_ },
    { (const float*)d_in[10], Whr, H_ * H_ },
    { (const float*)d_in[14], Whi, H_ * H_ },
    { (const float*)d_in[18], Whn, H_ * H_ },
    { (const float*)d_in[20], Wo1, H_ * H_ },
    { (const float*)d_in[22], Wo2, H_ * H_ },
    { (const float*)d_in[24], Wo3, DOUT_ * H_ },
    { inputs, x16, B_ * T_ * N_ * DIN_ },
  };
  for (int i = 0; i < 12; ++i)
    cvt_f32_f16_kernel<<<(cv[i].n + 255) / 256, 256, 0, stream>>>(cv[i].s, cv[i].d, cv[i].n);

  tsum_kernel<<<(B_ * E_ + 255) / 256, 256, 0, stream>>>(rel_types, ts, B_ * E_);

  hipMemsetAsync(h32,  0, (size_t)B_ * N_ * H_ * 4, stream);
  hipMemsetAsync(h16p, 0, (size_t)B_ * N_ * H_ * 2, stream);

  const float* b1  = (const float*)d_in[5];
  const float* b2  = (const float*)d_in[7];
  const float* bir = (const float*)d_in[9];
  const float* bhr = (const float*)d_in[11];
  const float* bii = (const float*)d_in[13];
  const float* bhi = (const float*)d_in[15];
  const float* bin_ = (const float*)d_in[17];
  const float* bhn = (const float*)d_in[19];
  const float* bo1 = (const float*)d_in[21];
  const float* bo2 = (const float*)d_in[23];
  const float* bo3 = (const float*)d_in[25];

  for (int t = 0; t < TS_; ++t) {
    msg_agg_kernel<<<B_ * N_, 128, SMEM_A, stream>>>(h16p, ts, W1h, b1, W2h, b2, agg);
    gru_out_kernel<<<(B_ * N_) / 16, 128, SMEM_B, stream>>>(
        x16, t, agg, h32, h16p,
        Wir, Wii, Win, Whr, Whi, Whn, Wo1, Wo2, Wo3,
        bir, bii, bin_, bhr, bhi, bhn, bo1, bo2, bo3,
        (float*)d_out);
  }
}
